// GraphTransformerWithYAtomMapPosEmb_36481452212248
// MI455X (gfx1250) — matrix operand order repaired and checked by reading, not compiled, not run on hardware
//
#include <hip/hip_runtime.h>

// ---------------- model dimensions ----------------
#define BSZ   4
#define NN    192
#define DXC   256
#define DEC   64
#define DYC   64
#define HMX   256
#define HME   128
#define HMY   128
#define FFXC  256
#define FFEC  128
#define FFYC  2048
#define INXC  40
#define INEC  10
#define INYC  13
#define OXC   17
#define OEC   5
#define NLAY  5

// ---------------- WMMA bf16 plumbing (CDNA5 / gfx1250, wave32) ----------------
typedef __attribute__((ext_vector_type(16))) __bf16 v16bf;
typedef __attribute__((ext_vector_type(8)))  float  v8f;
typedef __attribute__((ext_vector_type(4)))  int    v4i_;

union BF16Frag { v16bf v; unsigned short u[16]; unsigned int w32[8]; };

__device__ __forceinline__ unsigned short f2bf(float f) {
  unsigned u = __float_as_uint(f);
  u += 0x7fffu + ((u >> 16) & 1u);   // round-to-nearest-even
  return (unsigned short)(u >> 16);
}

__device__ __forceinline__ v8f wmma_bf16(const BF16Frag& a, const BF16Frag& b, v8f c) {
  return __builtin_amdgcn_wmma_f32_16x16x32_bf16(false, a.v, false, b.v, (short)0, c, false, false);
}

__device__ __forceinline__ v8f vzero8() {
  v8f z;
#pragma unroll
  for (int i = 0; i < 8; ++i) z[i] = 0.f;
  return z;
}

__device__ __forceinline__ void pack8(unsigned short* d, const float4& x, const float4& y) {
  d[0] = f2bf(x.x); d[1] = f2bf(x.y); d[2] = f2bf(x.z); d[3] = f2bf(x.w);
  d[4] = f2bf(y.x); d[5] = f2bf(y.y); d[6] = f2bf(y.z); d[7] = f2bf(y.w);
}

// modulated pack: d[j] = bf16(ya[c0+j] + ym[c0+j]*yr[c0+j])
__device__ __forceinline__ void pack8mod(unsigned short* d, const float* yr,
                                         const float* ya, const float* ym, int c0) {
#pragma unroll
  for (int j = 0; j < 8; ++j) d[j] = f2bf(ya[c0 + j] + ym[c0 + j] * yr[c0 + j]);
}

// ---- CDNA5 async global->LDS copy (16B per lane), with safe fallback ----
__device__ __forceinline__ void async_copy_f32x4(const float* g, float* l) {
#if defined(__gfx1250__) && __has_builtin(__builtin_amdgcn_global_load_async_to_lds_b128)
  typedef v4i_ __attribute__((address_space(1)))* gp_t;   // global (AS1) v4i*
  typedef v4i_ __attribute__((address_space(3)))* lp_t;   // LDS    (AS3) v4i*
  __builtin_amdgcn_global_load_async_to_lds_b128((gp_t)g, (lp_t)l, 0, 0);
#else
  *(float4*)l = *(const float4*)g;
#endif
}

__device__ __forceinline__ void async_wait() {
#if defined(__gfx1250__) && __has_builtin(__builtin_amdgcn_s_wait_asynccnt)
  __builtin_amdgcn_s_wait_asynccnt(0);
#elif defined(__gfx1250__)
  asm volatile("s_wait_asynccnt 0" ::: "memory");
#endif
}

// ---------------- guard-free fast GEMM (dominant shapes) ----------------
// Requires M%32==0 && N%64==0 (K tail handled with zero-fill).
// Block = 128 threads (4 waves); block tile 32x64; each wave: 16x32 out (2 WMMAs/step).
__global__ void __launch_bounds__(128) k_gemm_fast(
    const float* __restrict__ A, const float* __restrict__ W,
    const float* __restrict__ bias, float* __restrict__ C,
    int M, int N, int K, int act, int accum)
{
  __shared__ float          Af[32 * 32];   // A tile (f32, async-staged)
  __shared__ unsigned short Bt[64 * 32];   // B tile transposed [col][k] bf16
  const int tid  = threadIdx.x;
  const int wave = tid >> 5;
  const int lane = tid & 31;
  const int hlf  = lane >> 4;
  const int lc   = lane & 15;
  const int m0   = blockIdx.y * 32;
  const int nblk = blockIdx.x * 64;
  const int mrow = (wave & 1) * 16;
  const int ncol = (wave >> 1) * 32;
  v8f acc0 = vzero8(), acc1 = vzero8();
  for (int k0 = 0; k0 < K; k0 += 32) {
    const bool full = (k0 + 32 <= K);
    // stage A 32x32 (async b128 per lane when no K-tail)
    if (full) {
#pragma unroll
      for (int it = 0; it < 2; ++it) {
        int idx = tid + 128 * it;
        int r = idx >> 3, seg = idx & 7;
        async_copy_f32x4(&A[(size_t)(m0 + r) * K + k0 + seg * 4], &Af[r * 32 + seg * 4]);
      }
    } else {
#pragma unroll
      for (int it = 0; it < 2; ++it) {
        int idx = tid + 128 * it;
        int r = idx >> 3, seg = idx & 7;
#pragma unroll
        for (int e = 0; e < 4; ++e) {
          int gk = k0 + seg * 4 + e;
          Af[r * 32 + seg * 4 + e] = (gk < K) ? A[(size_t)(m0 + r) * K + gk] : 0.f;
        }
      }
    }
    // stage B 32x64 -> transposed bf16 (row-level K guard only; N is aligned)
#pragma unroll
    for (int it = 0; it < 4; ++it) {
      int idx = tid + 128 * it;
      int kk = idx >> 4, seg = idx & 15;
      int gk = k0 + kk;
      float4 v;
      if (gk < K) v = *(const float4*)&W[(size_t)gk * N + nblk + seg * 4];
      else { v.x = 0.f; v.y = 0.f; v.z = 0.f; v.w = 0.f; }
      Bt[(seg * 4 + 0) * 32 + kk] = f2bf(v.x);
      Bt[(seg * 4 + 1) * 32 + kk] = f2bf(v.y);
      Bt[(seg * 4 + 2) * 32 + kk] = f2bf(v.z);
      Bt[(seg * 4 + 3) * 32 + kk] = f2bf(v.w);
    }
    if (full && k0 + 32 < K) {   // prefetch next tiles (global_prefetch_b8)
      __builtin_prefetch(&A[(size_t)(m0 + (tid >> 3)) * K + k0 + 32 + (tid & 7) * 4], 0, 1);
      __builtin_prefetch(&W[(size_t)(k0 + 32 + (tid >> 4)) * N + nblk + (tid & 15) * 4], 0, 1);
    }
    async_wait();
    __syncthreads();
    BF16Frag a;
    {
      const float4* ar = (const float4*)&Af[(mrow + lc) * 32 + hlf * 8];
      pack8(&a.u[0], ar[0], ar[1]);
      pack8(&a.u[8], ar[4], ar[5]);
    }
#pragma unroll
    for (int nt = 0; nt < 2; ++nt) {
      BF16Frag bfr;
      const uint4* bp = (const uint4*)&Bt[(ncol + nt * 16 + lc) * 32 + hlf * 8];
      *(uint4*)&bfr.w32[0] = bp[0];
      *(uint4*)&bfr.w32[4] = bp[2];
      if (nt == 0) acc0 = wmma_bf16(a, bfr, acc0);
      else         acc1 = wmma_bf16(a, bfr, acc1);
    }
    __syncthreads();
  }
#pragma unroll
  for (int nt = 0; nt < 2; ++nt) {
    const v8f acc = nt ? acc1 : acc0;
    const int gn = nblk + ncol + nt * 16 + lc;
    const float bv = bias[gn];
#pragma unroll
    for (int i = 0; i < 8; ++i) {
      const int gm = m0 + mrow + i + 8 * hlf;
      float v = acc[i] + bv;
      if (act == 1) v = fmaxf(v, 0.f);
      else if (act == 2) v += 1.f;
      size_t o = (size_t)gm * N + gn;
      C[o] = accum ? (C[o] + v) : v;
    }
  }
}

// ---------------- guarded GEMM (small/tail shapes: M=4 rows, N=17/5) ----------------
__global__ void k_gemm(const float* __restrict__ A, const float* __restrict__ W,
                       const float* __restrict__ bias, float* __restrict__ C,
                       int M, int N, int K, int act, int accum)
{
  __shared__ unsigned short aT[16 * 32];
  __shared__ unsigned short bT[4][32 * 16];
  const int wave = threadIdx.x >> 5;
  const int lane = threadIdx.x & 31;
  const int hlf  = lane >> 4;
  const int lc   = lane & 15;
  const int m0 = blockIdx.y * 16;
  const int n0 = blockIdx.x * 64 + wave * 16;
  v8f acc = vzero8();
  for (int k0 = 0; k0 < K; k0 += 32) {
    for (int i = threadIdx.x; i < 16 * 32; i += 128) {
      int r = i >> 5, kk = i & 31;
      int gm = m0 + r, gk = k0 + kk;
      float v = (gm < M && gk < K) ? A[(size_t)gm * K + gk] : 0.f;
      aT[i] = f2bf(v);
    }
    for (int i = lane; i < 32 * 16; i += 32) {
      int kk = i >> 4, c = i & 15;
      int gk = k0 + kk, gn = n0 + c;
      float v = (gk < K && gn < N) ? W[(size_t)gk * N + gn] : 0.f;
      bT[wave][i] = f2bf(v);
    }
    __syncthreads();
    BF16Frag a, b;
#pragma unroll
    for (int j = 0; j < 8; ++j) {
      a.u[j]     = aT[lc * 32 + hlf * 8 + j];
      a.u[8 + j] = aT[lc * 32 + 16 + hlf * 8 + j];
    }
#pragma unroll
    for (int t = 0; t < 16; ++t) b.u[t] = bT[wave][((t < 8 ? 0 : 8) + hlf * 8 + t) * 16 + lc];
    acc = wmma_bf16(a, b, acc);
    __syncthreads();
  }
  const int gn = n0 + lc;
  if (gn < N) {
    const float bv = bias ? bias[gn] : 0.f;
#pragma unroll
    for (int i = 0; i < 8; ++i) {
      int gm = m0 + i + 8 * hlf;
      if (gm < M) {
        float v = acc[i] + bv;
        if (act == 1) v = fmaxf(v, 0.f);
        else if (act == 2) v += 1.f;
        size_t o = (size_t)gm * N + gn;
        C[o] = accum ? (C[o] + v) : v;
      }
    }
  }
}

// ---------------- fused edge/attention kernel ----------------
// One workgroup per (b, q): Y[k, c] slab (192*256 f32 = 196KB) lives in LDS
// (possible because CDNA5 allows up to 320KB per workgroup).
// Weight fragments are loop-invariant per workgroup -> hoisted into registers.
__global__ void __launch_bounds__(256) k_edge_attn(
    const float* __restrict__ E,
    const float* __restrict__ Q, const float* __restrict__ Kt, const float* __restrict__ V,
    const float* __restrict__ Wmul, const float* __restrict__ bmul,
    const float* __restrict__ Wadd, const float* __restrict__ badd,
    const float* __restrict__ Wout, const float* __restrict__ bout,
    const float* __restrict__ yEmul1, const float* __restrict__ yEadd,
    const float* __restrict__ yXmul1, const float* __restrict__ yXadd,
    const int* __restrict__ nmask,
    float* __restrict__ newE, float* __restrict__ newXpre)
{
  extern __shared__ float smem[];
  float* Ylds = smem;                    // NN*DXC
  float* qrow = Ylds + NN * DXC;         // DXC
  float* kmsk = qrow + DXC;              // NN
  float* yem  = kmsk + NN;               // DXC
  float* yea  = yem + DXC;               // DXC
  const int b   = blockIdx.y;
  const int qi  = blockIdx.x;
  const int tid = threadIdx.x;
  const int lane = tid & 31;
  const int wave = tid >> 5;
  const int hlf  = lane >> 4;
  const int lc   = lane & 15;
  for (int i = tid; i < DXC; i += 256) {
    qrow[i] = Q[((size_t)b * NN + qi) * DXC + i];
    yem[i]  = yEmul1[b * DXC + i];
    yea[i]  = yEadd[b * DXC + i];
  }
  for (int i = tid; i < NN; i += 256) kmsk[i] = nmask[b * NN + i] ? 1.f : 0.f;
  __syncthreads();
  const float mq = kmsk[qi];
  const float scale = 0.17677669529663689f;  // 1/sqrt(DF=32)
  const size_t erowbase = ((size_t)b * NN + qi) * NN;

  // ---- phase 1: Y = Q*K/sqrt(DF) * (E@Wmul+bm+1) + (E@Wadd+ba), masked ----
  {
    const int n0 = wave * 32;
    BF16Frag Bm[2][2], Ba[2][2];           // hoisted weight fragments
#pragma unroll
    for (int kk2 = 0; kk2 < 2; ++kk2)
#pragma unroll
      for (int nt = 0; nt < 2; ++nt) {
        const int c = n0 + nt * 16 + lc;
#pragma unroll
        for (int t = 0; t < 16; ++t) {
          const int kr = kk2 * 32 + (t < 8 ? 0 : 8) + hlf * 8 + t;
          Bm[kk2][nt].u[t] = f2bf(Wmul[(size_t)kr * DXC + c]);
          Ba[kk2][nt].u[t] = f2bf(Wadd[(size_t)kr * DXC + c]);
        }
      }
    for (int kt = 0; kt < NN / 16; ++kt) {
      const int k0 = kt * 16;
      if (k0 + 16 < NN)
        __builtin_prefetch(&E[(erowbase + (size_t)(k0 + 16 + lc)) * DEC], 0, 1);
      v8f am[2], aa[2];
      am[0] = vzero8(); am[1] = vzero8(); aa[0] = vzero8(); aa[1] = vzero8();
#pragma unroll
      for (int kk2 = 0; kk2 < 2; ++kk2) {
        const float* er = &E[(erowbase + (size_t)(k0 + lc)) * DEC + kk2 * 32];
        BF16Frag af;
        float4 x0 = *(const float4*)(er + hlf * 8);
        float4 x1 = *(const float4*)(er + hlf * 8 + 4);
        float4 x2 = *(const float4*)(er + 16 + hlf * 8);
        float4 x3 = *(const float4*)(er + 16 + hlf * 8 + 4);
        pack8(&af.u[0], x0, x1);
        pack8(&af.u[8], x2, x3);
#pragma unroll
        for (int nt = 0; nt < 2; ++nt) {
          am[nt] = wmma_bf16(af, Bm[kk2][nt], am[nt]);
          aa[nt] = wmma_bf16(af, Ba[kk2][nt], aa[nt]);
        }
      }
#pragma unroll
      for (int nt = 0; nt < 2; ++nt) {
        const int c = n0 + nt * 16 + lc;
        const float bm_ = bmul[c], ba_ = badd[c];
        const float qv = qrow[c] * scale;
#pragma unroll
        for (int i = 0; i < 8; ++i) {
          const int kr = k0 + i + 8 * hlf;
          const float mk = kmsk[kr] * mq;
          const float e1 = (am[nt][i] + bm_) * mk;
          const float e2 = (aa[nt][i] + ba_) * mk;
          const float kv = Kt[((size_t)b * NN + kr) * DXC + c];
          Ylds[kr * DXC + c] = qv * kv * (e1 + 1.f) + e2;
        }
      }
    }
  }
  __syncthreads();

  // ---- phase 2: newE = ((yea + yem*Y) @ Wout + bout) * mask ----
  {
    const int nt2 = wave & 3;
    const int n0 = nt2 * 16;
    const int c = n0 + lc;
    BF16Frag Bo[8];                        // hoisted Wout fragments (K=256)
#pragma unroll
    for (int kk2 = 0; kk2 < 8; ++kk2)
#pragma unroll
      for (int t = 0; t < 16; ++t)
        Bo[kk2].u[t] = f2bf(Wout[(size_t)(kk2 * 32 + (t < 8 ? 0 : 8) + hlf * 8 + t) * DEC + c]);
    const int ktbase = (wave >> 2) * 6;    // waves 0-3: kt 0..5, waves 4-7: kt 6..11
    for (int kti = 0; kti < 6; ++kti) {
      const int k0 = (ktbase + kti) * 16;
      v8f acc = vzero8();
      const float* yr = &Ylds[(k0 + lc) * DXC];
#pragma unroll
      for (int kk2 = 0; kk2 < 8; ++kk2) {
        BF16Frag af;
        const int c0 = kk2 * 32 + hlf * 8;
        pack8mod(&af.u[0], yr, yea, yem, c0);
        pack8mod(&af.u[8], yr, yea, yem, c0 + 16);
        acc = wmma_bf16(af, Bo[kk2], acc);
      }
      const float bo = bout[c];
#pragma unroll
      for (int i = 0; i < 8; ++i) {
        const int kr = k0 + i + 8 * hlf;
        const float mk = kmsk[kr] * mq;
        newE[(erowbase + kr) * DEC + c] = (acc[i] + bo) * mk;
      }
    }
  }

  // ---- phase 3: per-channel softmax over k + attn@V ----
  {
    const int c = tid;                     // 256 threads = 256 channels
    float mx = -INFINITY;
    for (int k = 0; k < NN; ++k)
      if (kmsk[k] > 0.f) mx = fmaxf(mx, Ylds[k * DXC + c]);
    float den = 0.f, num = 0.f;
    for (int k = 0; k < NN; ++k) {
      if (kmsk[k] > 0.f) {
        const float e = __expf(Ylds[k * DXC + c] - mx);
        den += e;
        num += e * V[((size_t)b * NN + k) * DXC + c];
      }
    }
    const float wv = (den > 0.f) ? (num / den) : 0.f;
    newXpre[((size_t)b * NN + qi) * DXC + c] =
        yXadd[b * DXC + c] + yXmul1[b * DXC + c] * wv;
  }
}

// ---------------- elementwise / reduction kernels ----------------
__global__ void k_mask_rows(float* __restrict__ X, const int* __restrict__ m, int total, int D) {
  int i = blockIdx.x * 256 + threadIdx.x;
  if (i >= total) return;
  if (!m[i / D]) X[i] = 0.f;
}

__global__ void k_add_inplace(float* __restrict__ a, const float* __restrict__ b, int n) {
  int i = blockIdx.x * 256 + threadIdx.x;
  if (i < n) a[i] += b[i];
}

__global__ void k_sym_mask(const float* __restrict__ In, const int* __restrict__ m,
                           float* __restrict__ Out, int D, int use_mask) {
  size_t idx = (size_t)blockIdx.x * 256 + threadIdx.x;
  size_t total = (size_t)BSZ * NN * NN * D;
  if (idx >= total) return;
  int c = (int)(idx % D);
  size_t e = idx / D;
  int j = (int)(e % NN);
  size_t t = e / NN;
  int i = (int)(t % NN);
  int b = (int)(t / NN);
  float v = 0.5f * (In[idx] + In[(((size_t)b * NN + j) * NN + i) * D + c]);
  if (use_mask && !(m[b * NN + i] && m[b * NN + j])) v = 0.f;
  Out[idx] = v;
}

// out = LayerNorm(X + R) * S + B ; one wave per row (D in {64, 256})
__global__ void k_add_ln(const float* __restrict__ X, const float* __restrict__ R,
                         const float* __restrict__ S, const float* __restrict__ B,
                         float* __restrict__ out, int rows, int D)
{
  const int wave = threadIdx.x >> 5, lane = threadIdx.x & 31;
  const int row = blockIdx.x * 8 + wave;
  if (row >= rows) return;
  const float* xr = X + (size_t)row * D;
  const float* rr = R + (size_t)row * D;
  float vals[8];
  const int cnt = D / 32;
  float s1 = 0.f, s2 = 0.f;
  for (int i = 0; i < cnt; ++i) {
    int d = lane + 32 * i;
    float v = xr[d] + rr[d];
    vals[i] = v; s1 += v; s2 += v * v;
  }
  for (int o = 16; o > 0; o >>= 1) { s1 += __shfl_xor(s1, o); s2 += __shfl_xor(s2, o); }
  const float mean = s1 / D;
  const float var  = s2 / D - mean * mean;
  const float inv  = rsqrtf(var + 1e-5f);
  for (int i = 0; i < cnt; ++i) {
    int d = lane + 32 * i;
    out[(size_t)row * D + d] = (vals[i] - mean) * inv * S[d] + B[d];
  }
}

// xz = concat(mean, min, max, std(ddof=1)) over nodes -> (BSZ, 4*DXC)
__global__ void k_stats_X(const float* __restrict__ Xh, float* __restrict__ xz) {
  const int b = blockIdx.x, d = threadIdx.x;
  float s = 0.f, s2 = 0.f, mn = INFINITY, mx = -INFINITY;
  for (int i = 0; i < NN; ++i) {
    float v = Xh[((size_t)b * NN + i) * DXC + d];
    s += v; s2 += v * v; mn = fminf(mn, v); mx = fmaxf(mx, v);
  }
  const float mean = s / NN;
  const float var  = (s2 - s * s / NN) / (NN - 1);
  xz[b * 4 * DXC + d]           = mean;
  xz[b * 4 * DXC + DXC + d]     = mn;
  xz[b * 4 * DXC + 2 * DXC + d] = mx;
  xz[b * 4 * DXC + 3 * DXC + d] = sqrtf(fmaxf(var, 0.f));
}

// ez over (n,n) -> (BSZ, 4*DEC); block per (b,d)
__global__ void k_stats_E(const float* __restrict__ Eh, float* __restrict__ ez) {
  __shared__ float rs[256], rs2[256], rmn[256], rmx[256];
  const int b = blockIdx.x / DEC, d = blockIdx.x % DEC;
  float s = 0.f, s2 = 0.f, mn = INFINITY, mx = -INFINITY;
  for (int i = threadIdx.x; i < NN * NN; i += 256) {
    float v = Eh[((size_t)b * NN * NN + i) * DEC + d];
    s += v; s2 += v * v; mn = fminf(mn, v); mx = fmaxf(mx, v);
  }
  rs[threadIdx.x] = s; rs2[threadIdx.x] = s2; rmn[threadIdx.x] = mn; rmx[threadIdx.x] = mx;
  __syncthreads();
  for (int o = 128; o > 0; o >>= 1) {
    if (threadIdx.x < o) {
      rs[threadIdx.x]  += rs[threadIdx.x + o];
      rs2[threadIdx.x] += rs2[threadIdx.x + o];
      rmn[threadIdx.x] = fminf(rmn[threadIdx.x], rmn[threadIdx.x + o]);
      rmx[threadIdx.x] = fmaxf(rmx[threadIdx.x], rmx[threadIdx.x + o]);
    }
    __syncthreads();
  }
  if (threadIdx.x == 0) {
    const float n = (float)(NN * NN);
    const float mean = rs[0] / n;
    const float var  = (rs2[0] - rs[0] * rs[0] / n) / (n - 1.f);
    ez[b * 4 * DEC + d]           = mean;
    ez[b * 4 * DEC + DEC + d]     = rmn[0];
    ez[b * 4 * DEC + 2 * DEC + d] = rmx[0];
    ez[b * 4 * DEC + 3 * DEC + d] = sqrtf(fmaxf(var, 0.f));
  }
}

// ---------------- output assembly ----------------
__global__ void k_out_X(const float* __restrict__ T, const float* __restrict__ Xin,
                        const int* __restrict__ m, float* __restrict__ out) {
  int idx = blockIdx.x * 256 + threadIdx.x;
  if (idx >= BSZ * NN * OXC) return;
  int c = idx % OXC; int r = idx / OXC;
  float v = T[(size_t)r * OXC + c] + Xin[(size_t)r * INXC + c];
  out[idx] = m[r] ? v : 0.f;
}

__global__ void k_out_E(const float* __restrict__ T, const float* __restrict__ Ein,
                        const int* __restrict__ m, float* __restrict__ out) {
  size_t idx = (size_t)blockIdx.x * 256 + threadIdx.x;
  size_t total = (size_t)BSZ * NN * NN * OEC;
  if (idx >= total) return;
  int c = (int)(idx % OEC);
  size_t e = idx / OEC;
  int j = (int)(e % NN);
  size_t t = e / NN;
  int i = (int)(t % NN);
  int b = (int)(t / NN);
  size_t ij = ((size_t)b * NN + i) * NN + j;
  size_t ji = ((size_t)b * NN + j) * NN + i;
  float a = T[ij * OEC + c] + Ein[ij * INEC + c];
  float s = T[ji * OEC + c] + Ein[ji * INEC + c];
  float v = (i != j) ? 0.5f * (a + s) : 0.f;
  if (!(m[b * NN + i] && m[b * NN + j])) v = 0.f;
  out[idx] = v;
}

__global__ void k_out_y(const float* __restrict__ yh, const float* __restrict__ yin,
                        float* __restrict__ out) {
  int b = threadIdx.x;
  if (b < BSZ) out[b] = yh[b * DYC] + yin[b * INYC];
}

// ---------------- parameter layout (jax pytree: sorted-dict flatten) ----------------
enum {
  L_e_add_b = 0, L_e_add_w, L_e_mul_b, L_e_mul_w, L_e_out_b, L_e_out_w,
  L_e_y_b, L_e_y_w, L_k_b, L_k_w,
  L_linE1_b, L_linE1_w, L_linE2_b, L_linE2_w,
  L_linX1_b, L_linX1_w, L_linX2_b, L_linX2_w,
  L_liny1_b, L_liny1_w, L_liny2_b, L_liny2_w,
  L_lnE1_b, L_lnE1_s, L_lnE2_b, L_lnE2_s,
  L_lnX1_b, L_lnX1_s, L_lnX2_b, L_lnX2_s,
  L_lny1_b, L_lny1_s, L_lny2_b, L_lny2_s,
  L_q_b, L_q_w, L_v_b, L_v_w,
  L_x_out_b, L_x_out_w, L_x_y_b, L_x_y_w,
  L_y_e_add_b, L_y_e_add_w, L_y_e_mul_b, L_y_e_mul_w,
  L_y_out1_b, L_y_out1_w, L_y_out2_b, L_y_out2_w,
  L_y_x_add_b, L_y_x_add_w, L_y_x_mul_b, L_y_x_mul_w,
  L_y_y_b, L_y_y_w          // 56 leaves per layer
};

static inline int cdiv(int a, int b) { return (a + b - 1) / b; }

extern "C" void kernel_launch(void* const* d_in, const int* in_sizes, int n_in,
                              void* d_out, int out_size, void* d_ws, size_t ws_size,
                              hipStream_t stream)
{
  (void)in_sizes; (void)n_in; (void)out_size; (void)ws_size;
  const float* Xin  = (const float*)d_in[0];
  const float* Ein  = (const float*)d_in[1];
  const float* yin  = (const float*)d_in[2];
  const int*   nmsk = (const int*)d_in[3];
  const float* pos  = (const float*)d_in[5];
  const int P0 = 7;   // params start after 7 data inputs
  auto pw = [&](int i) { return (const float*)d_in[i]; };

  // workspace carve-up (fp32), total ~157MB
  float* w = (float*)d_ws;
  auto take = [&](size_t n) { float* p = w; w += n; return p; };
  float* Xh  = take((size_t)BSZ * NN * DXC);
  float* Qb  = take((size_t)BSZ * NN * DXC);
  float* Kb  = take((size_t)BSZ * NN * DXC);
  float* Vb  = take((size_t)BSZ * NN * DXC);
  float* Xt  = take((size_t)BSZ * NN * DXC);
  float* hX  = take((size_t)BSZ * NN * DXC);
  float* nXp = take((size_t)BSZ * NN * DXC);
  float* Eh  = take((size_t)BSZ * NN * NN * DEC);
  float* EnE = take((size_t)BSZ * NN * NN * DEC);
  float* hE  = take((size_t)BSZ * NN * NN * HME);
  float* xz  = take((size_t)BSZ * 4 * DXC);
  float* ez  = take((size_t)BSZ * 4 * DEC);
  float* yh  = take((size_t)BSZ * DYC);
  float* yt1 = take((size_t)BSZ * DYC);
  float* yt2 = take((size_t)BSZ * DYC);
  float* nyb = take((size_t)BSZ * DYC);
  float* hy  = take((size_t)BSZ * FFYC);
  float* yEm = take((size_t)BSZ * DXC);
  float* yEa = take((size_t)BSZ * DXC);
  float* yXm = take((size_t)BSZ * DXC);
  float* yXa = take((size_t)BSZ * DXC);

  auto gemm = [&](const float* A, const float* W_, const float* bias, float* C,
                  int M, int N, int K, int act, int accum) {
    const bool fast = (M % 32 == 0) && (N % 64 == 0) && (K % 4 == 0 || K < 32);
    if (fast) {
      dim3 g(N / 64, M / 32);
      k_gemm_fast<<<g, dim3(128), 0, stream>>>(A, W_, bias, C, M, N, K, act, accum);
    } else {
      dim3 g(cdiv(N, 64), cdiv(M, 16));
      k_gemm<<<g, dim3(128), 0, stream>>>(A, W_, bias, C, M, N, K, act, accum);
    }
  };
  const int XROWS = BSZ * NN;        // 768
  const int EROWS = BSZ * NN * NN;   // 147456

  // ---------------- input blocks ----------------
  gemm(Ein, pw(P0 + 1), pw(P0 + 0), hE, EROWS, HME, INEC, 1, 0);      // in_E1 + relu
  gemm(hE, pw(P0 + 3), pw(P0 + 2), EnE, EROWS, DEC, HME, 1, 0);       // in_E2 + relu
  {
    size_t tot = (size_t)EROWS * DEC;
    k_sym_mask<<<(unsigned)((tot + 255) / 256), 256, 0, stream>>>(EnE, nmsk, Eh, DEC, 1);
  }
  gemm(Xin, pw(P0 + 5), pw(P0 + 4), hX, XROWS, HMX, INXC, 1, 0);      // in_X1 + relu
  gemm(hX, pw(P0 + 7), pw(P0 + 6), Xh, XROWS, DXC, HMX, 1, 0);        // in_X2 + relu
  k_mask_rows<<<cdiv(XROWS * DXC, 256), 256, 0, stream>>>(Xh, nmsk, XROWS * DXC, DXC);
  k_add_inplace<<<cdiv(XROWS * DXC, 256), 256, 0, stream>>>(Xh, pos, XROWS * DXC);
  gemm(yin, pw(P0 + 9), pw(P0 + 8), hy, BSZ, HMY, INYC, 1, 0);        // in_y1 + relu
  gemm(hy, pw(P0 + 11), pw(P0 + 10), yh, BSZ, DYC, HMY, 1, 0);        // in_y2 + relu

  const size_t edge_smem = (size_t)(NN * DXC + DXC + NN + DXC + DXC) * sizeof(float);

  // ---------------- transformer layers ----------------
  for (int l = 0; l < NLAY; ++l) {
    const int LB = P0 + 12 + l * 56;
    auto LW = [&](int off) { return (const float*)d_in[LB + off]; };
    gemm(Xh, LW(L_q_w), LW(L_q_b), Qb, XROWS, DXC, DXC, 0, 0);
    gemm(Xh, LW(L_k_w), LW(L_k_b), Kb, XROWS, DXC, DXC, 0, 0);
    gemm(Xh, LW(L_v_w), LW(L_v_b), Vb, XROWS, DXC, DXC, 0, 0);
    k_mask_rows<<<cdiv(XROWS * DXC, 256), 256, 0, stream>>>(Qb, nmsk, XROWS * DXC, DXC);
    k_mask_rows<<<cdiv(XROWS * DXC, 256), 256, 0, stream>>>(Kb, nmsk, XROWS * DXC, DXC);
    k_mask_rows<<<cdiv(XROWS * DXC, 256), 256, 0, stream>>>(Vb, nmsk, XROWS * DXC, DXC);
    // y-conditioned modulation vectors (act=2 folds the "+1.0")
    gemm(yh, LW(L_y_e_mul_w), LW(L_y_e_mul_b), yEm, BSZ, DXC, DYC, 2, 0);
    gemm(yh, LW(L_y_e_add_w), LW(L_y_e_add_b), yEa, BSZ, DXC, DYC, 0, 0);
    gemm(yh, LW(L_y_x_mul_w), LW(L_y_x_mul_b), yXm, BSZ, DXC, DYC, 2, 0);
    gemm(yh, LW(L_y_x_add_w), LW(L_y_x_add_b), yXa, BSZ, DXC, DYC, 0, 0);
    // fused edge / per-channel attention block
    k_edge_attn<<<dim3(NN, BSZ), dim3(256), edge_smem, stream>>>(
        Eh, Qb, Kb, Vb,
        LW(L_e_mul_w), LW(L_e_mul_b), LW(L_e_add_w), LW(L_e_add_b),
        LW(L_e_out_w), LW(L_e_out_b),
        yEm, yEa, yXm, yXa, nmsk, EnE, nXp);
    gemm(nXp, LW(L_x_out_w), LW(L_x_out_b), Xt, XROWS, DXC, DXC, 0, 0);
    k_mask_rows<<<cdiv(XROWS * DXC, 256), 256, 0, stream>>>(Xt, nmsk, XROWS * DXC, DXC);
    // y path (pooled stats from pre-update Xh/Eh)
    k_stats_X<<<BSZ, DXC, 0, stream>>>(Xh, xz);
    k_stats_E<<<BSZ * DEC, 256, 0, stream>>>(Eh, ez);
    gemm(yh, LW(L_y_y_w), LW(L_y_y_b), yt1, BSZ, DYC, DYC, 0, 0);
    gemm(xz, LW(L_x_y_w), LW(L_x_y_b), yt1, BSZ, DYC, 4 * DXC, 0, 1);
    gemm(ez, LW(L_e_y_w), LW(L_e_y_b), yt1, BSZ, DYC, 4 * DEC, 0, 1);
    gemm(yt1, LW(L_y_out1_w), LW(L_y_out1_b), yt2, BSZ, DYC, DYC, 1, 0);
    gemm(yt2, LW(L_y_out2_w), LW(L_y_out2_b), nyb, BSZ, DYC, DYC, 0, 0);
    // X residual + FF
    k_add_ln<<<cdiv(XROWS, 8), 256, 0, stream>>>(Xh, Xt, LW(L_lnX1_s), LW(L_lnX1_b), Xh, XROWS, DXC);
    gemm(Xh, LW(L_linX1_w), LW(L_linX1_b), hX, XROWS, FFXC, DXC, 1, 0);
    gemm(hX, LW(L_linX2_w), LW(L_linX2_b), Xt, XROWS, DXC, FFXC, 0, 0);
    k_add_ln<<<cdiv(XROWS, 8), 256, 0, stream>>>(Xh, Xt, LW(L_lnX2_s), LW(L_lnX2_b), Xh, XROWS, DXC);
    // E residual + FF
    k_add_ln<<<cdiv(EROWS, 8), 256, 0, stream>>>(Eh, EnE, LW(L_lnE1_s), LW(L_lnE1_b), Eh, EROWS, DEC);
    gemm(Eh, LW(L_linE1_w), LW(L_linE1_b), hE, EROWS, FFEC, DEC, 1, 0);
    gemm(hE, LW(L_linE2_w), LW(L_linE2_b), EnE, EROWS, DEC, FFEC, 0, 0);
    k_add_ln<<<cdiv(EROWS, 8), 256, 0, stream>>>(Eh, EnE, LW(L_lnE2_s), LW(L_lnE2_b), Eh, EROWS, DEC);
    // y residual + FF
    k_add_ln<<<1, 256, 0, stream>>>(yh, nyb, LW(L_lny1_s), LW(L_lny1_b), yh, BSZ, DYC);
    gemm(yh, LW(L_liny1_w), LW(L_liny1_b), hy, BSZ, FFYC, DYC, 1, 0);
    gemm(hy, LW(L_liny2_w), LW(L_liny2_b), nyb, BSZ, DYC, FFYC, 0, 0);
    k_add_ln<<<1, 256, 0, stream>>>(yh, nyb, LW(L_lny2_s), LW(L_lny2_b), yh, BSZ, DYC);
  }

  // ---------------- output blocks ----------------
  float* out = (float*)d_out;
  const int OB = P0 + 12 + NLAY * 56;   // out_E1.b/w, out_E2.b/w, out_X1.b/w, out_X2.b/w
  gemm(Xh, pw(OB + 5), pw(OB + 4), hX, XROWS, HMX, DXC, 1, 0);
  gemm(hX, pw(OB + 7), pw(OB + 6), Xt, XROWS, OXC, HMX, 0, 0);
  k_out_X<<<cdiv(XROWS * OXC, 256), 256, 0, stream>>>(Xt, Xin, nmsk, out);
  gemm(Eh, pw(OB + 1), pw(OB + 0), hE, EROWS, HME, DEC, 1, 0);
  gemm(hE, pw(OB + 3), pw(OB + 2), EnE, EROWS, OEC, HME, 0, 0);
  k_out_E<<<cdiv(EROWS * OEC, 256), 256, 0, stream>>>(EnE, Ein, nmsk, out + (size_t)XROWS * OXC);
  k_out_y<<<1, BSZ, 0, stream>>>(yh, yin, out + (size_t)XROWS * OXC + (size_t)EROWS * OEC);
}
